// NeuronCircuit_41549513621649
// MI455X (gfx1250) — compile-verified, matrix operand order checked by reading, not run on hardware
//
#include <hip/hip_runtime.h>
#include <hip/hip_bf16.h>

typedef __attribute__((ext_vector_type(16))) _Float16 v16h;
typedef __attribute__((ext_vector_type(8)))  _Float16 v8h;
typedef __attribute__((ext_vector_type(8)))  float    v8f;

// ---------------------------------------------------------------------------
// WMMA helpers (CDNA5 / gfx1250, wave32)
// 16-bit A-fragment layout: lane holds row M=lane&15; element e holds
// K = (e&7) + (e>=8?16:0) + (lane>=16?8:0)  -> two contiguous 8-half chunks.
// B fragments use the mirrored layout (lane -> N), so we stage B as [N][K].
// ---------------------------------------------------------------------------
static __device__ __forceinline__ v16h load_frag(const _Float16* __restrict__ base,
                                                 int row, int pitch, int kbase, int lane) {
  const _Float16* p = base + row * pitch + kbase + ((lane & 16) ? 8 : 0);
  v8h lo = *(const v8h*)(p);
  v8h hi = *(const v8h*)(p + 16);
  return __builtin_shufflevector(lo, hi, 0,1,2,3,4,5,6,7,8,9,10,11,12,13,14,15);
}

static __device__ __forceinline__ v8f wmma32(v16h a, v16h b, v8f c) {
  return __builtin_amdgcn_wmma_f32_16x16x32_f16(false, a, false, b, (short)0, c, false, false);
}

// ---------------------------------------------------------------------------
// Generic batched GEMM: C[M,N] = A[M,K] * B[K,N]   (BT: B given as [N][K])
// fp32 inputs converted to f16 in LDS, f32 accumulate, OutT output.
// Block = 256 threads (8 waves), 64x64 C tile, K-step 64,
// software-pipelined double-buffered LDS (one barrier per K-step, global
// fetch of tile i+1 overlapped with WMMA on tile i).
// ---------------------------------------------------------------------------
template<bool BT, typename OutT>
__global__ __launch_bounds__(256)
void gemm_f16w(const float* __restrict__ A, const float* __restrict__ Bm,
               OutT* __restrict__ C, int M, int N, int K,
               int lda, int ldb, int ldc,
               long long sA, long long sB, long long sC)
{
  const int z = blockIdx.z;
  A  += (long long)z * sA;
  Bm += (long long)z * sB;
  C  += (long long)z * sC;
  const int bm = blockIdx.y * 64, bn = blockIdx.x * 64;
  __shared__ __attribute__((aligned(16))) _Float16 sAx[2][64 * 64];
  __shared__ __attribute__((aligned(16))) _Float16 sBx[2][64 * 64];
  const int tid = threadIdx.x, lane = tid & 31, wave = tid >> 5;
  const int wr = wave & 3, wc0 = wave >> 2, l15 = lane & 15;
  const int ar = tid >> 2, ac = (tid & 3) * 16;   // 64 rows x 64 cols, 16/thread

  float aReg[16], bReg[16];
  auto fetch = [&](int kk) {
    const int gm = bm + ar;
    #pragma unroll
    for (int i = 0; i < 16; ++i) {
      int gk = kk + ac + i;
      aReg[i] = (gm < M && gk < K) ? A[(long long)gm * lda + gk] : 0.f;
    }
    if (BT) {                 // B given [N][K]
      const int gn = bn + ar;
      #pragma unroll
      for (int i = 0; i < 16; ++i) {
        int gk = kk + ac + i;
        bReg[i] = (gn < N && gk < K) ? Bm[(long long)gn * ldb + gk] : 0.f;
      }
    } else {                  // B given [K][N]
      const int gk = kk + ar;
      #pragma unroll
      for (int i = 0; i < 16; ++i) {
        int gn = bn + ac + i;
        bReg[i] = (gk < K && gn < N) ? Bm[(long long)gk * ldb + gn] : 0.f;
      }
    }
  };
  auto stage = [&](int buf) {
    #pragma unroll
    for (int i = 0; i < 16; ++i) sAx[buf][ar * 64 + ac + i] = (_Float16)aReg[i];
    if (BT) {
      #pragma unroll
      for (int i = 0; i < 16; ++i) sBx[buf][ar * 64 + ac + i] = (_Float16)bReg[i];
    } else {                  // transpose into [N][K]
      #pragma unroll
      for (int i = 0; i < 16; ++i) sBx[buf][(ac + i) * 64 + ar] = (_Float16)bReg[i];
    }
  };

  v8f c0 = {}; v8f c1 = {};
  const int nsteps = (K + 63) >> 6;
  fetch(0);
  stage(0);
  __syncthreads();
  for (int s = 0; s < nsteps; ++s) {
    const int cur = s & 1;
    if (s + 1 < nsteps) fetch((s + 1) * 64);   // overlap with WMMA below
    #pragma unroll
    for (int kc = 0; kc < 64; kc += 32) {
      v16h a  = load_frag(sAx[cur], wr * 16 + l15, 64, kc, lane);
      v16h b0 = load_frag(sBx[cur], wc0 * 16 + l15, 64, kc, lane);
      v16h b1 = load_frag(sBx[cur], (wc0 + 2) * 16 + l15, 64, kc, lane);
      c0 = wmma32(a, b0, c0);
      c1 = wmma32(a, b1, c1);
    }
    if (s + 1 < nsteps) stage(cur ^ 1);
    __syncthreads();
  }

  const int rbase = bm + wr * 16 + ((lane & 16) ? 8 : 0);
  const int col0 = bn + wc0 * 16 + l15;
  const int col1 = col0 + 32;
  #pragma unroll
  for (int i = 0; i < 8; ++i) {
    int row = rbase + i;
    if (row < M) {
      if (col0 < N) C[(long long)row * ldc + col0] = (OutT)c0[i];
      if (col1 < N) C[(long long)row * ldc + col1] = (OutT)c1[i];
    }
  }
}

// ---------------------------------------------------------------------------
// Flash attention (causal): one block per (q-tile of 64 rows, head, batch).
// Q/K/V are f16 in [b][s][h*64+c] layout. O written f32, same layout.
// ---------------------------------------------------------------------------
__global__ __launch_bounds__(256)
void flash_attn(const _Float16* __restrict__ Qg, const _Float16* __restrict__ Kg,
                const _Float16* __restrict__ Vg, float* __restrict__ Og)
{
  const int S = 1024, D = 1024;
  const int qt = blockIdx.x, hI = blockIdx.y, b = blockIdx.z;
  __shared__ __attribute__((aligned(16))) _Float16 lQ [64 * 64];
  __shared__ __attribute__((aligned(16))) _Float16 lK [64 * 64];  // [kv][d]
  __shared__ __attribute__((aligned(16))) _Float16 lVt[64 * 64];  // [d][kv]
  __shared__ __attribute__((aligned(16))) _Float16 lP [64 * 64];  // [q][kv]
  __shared__ float lS[64 * 64];
  __shared__ float mrow[64], lrow[64], srow[64];
  const int tid = threadIdx.x, lane = tid & 31, wave = tid >> 5;
  const int wr = wave & 3, wc0 = wave >> 2, l15 = lane & 15;
  const long long rowQ0 = ((long long)b * S + qt * 64) * D + hI * 64;
  { // load Q tile, pre-scaled by 1/sqrt(dh)=0.125
    int r = tid >> 2, c = (tid & 3) * 16;
    #pragma unroll
    for (int i = 0; i < 16; ++i)
      lQ[r * 64 + c + i] = (_Float16)((float)Qg[rowQ0 + (long long)r * D + c + i] * 0.125f);
  }
  if (tid < 64) { mrow[tid] = -3.0e38f; lrow[tid] = 0.f; }
  v8f o0 = {}; v8f o1 = {};
  __syncthreads();
  for (int kt = 0; kt <= qt; ++kt) {
    const long long rowK0 = ((long long)b * S + kt * 64) * D + hI * 64;
    { int r = tid >> 2, c = (tid & 3) * 16;
      #pragma unroll
      for (int i = 0; i < 16; ++i) {
        int cc = c + i;
        _Float16 kv = Kg[rowK0 + (long long)r * D + cc];
        _Float16 vv = Vg[rowK0 + (long long)r * D + cc];
        lK [r * 64 + cc] = kv;
        lVt[cc * 64 + r] = vv;
      }
    }
    __syncthreads();
    // S = Q * K^T  (B fragments read lK as [n=kv][k=d])
    v8f s0 = {}; v8f s1 = {};
    #pragma unroll
    for (int kc = 0; kc < 64; kc += 32) {
      v16h a  = load_frag(lQ, wr * 16 + l15, 64, kc, lane);
      v16h b0 = load_frag(lK, wc0 * 16 + l15, 64, kc, lane);
      v16h b1 = load_frag(lK, (wc0 + 2) * 16 + l15, 64, kc, lane);
      s0 = wmma32(a, b0, s0);
      s1 = wmma32(a, b1, s1);
    }
    { int rb = wr * 16 + ((lane & 16) ? 8 : 0);
      #pragma unroll
      for (int i = 0; i < 8; ++i) {
        lS[(rb + i) * 64 + wc0 * 16 + l15]      = s0[i];
        lS[(rb + i) * 64 + wc0 * 16 + 32 + l15] = s1[i];
      }
    }
    __syncthreads();
    if (tid < 64) { // online softmax, one row per thread
      const int r = tid, grow = qt * 64 + r;
      float mprev = mrow[r];
      float mx = mprev;
      for (int j = 0; j < 64; ++j) {
        float v = (kt * 64 + j <= grow) ? lS[r * 64 + j] : -3.0e38f;
        mx = fmaxf(mx, v);
      }
      float sc = __expf(mprev - mx);
      float sum = 0.f;
      for (int j = 0; j < 64; ++j) {
        float v = (kt * 64 + j <= grow) ? lS[r * 64 + j] : -3.0e38f;
        float p = __expf(v - mx);
        lP[r * 64 + j] = (_Float16)p;
        sum += p;
      }
      mrow[r] = mx;
      lrow[r] = lrow[r] * sc + sum;
      srow[r] = sc;
    }
    __syncthreads();
    { // rescale O accumulators, then O += P * V
      int rb = wr * 16 + ((lane & 16) ? 8 : 0);
      #pragma unroll
      for (int i = 0; i < 8; ++i) { float f = srow[rb + i]; o0[i] *= f; o1[i] *= f; }
    }
    #pragma unroll
    for (int kc = 0; kc < 64; kc += 32) {
      v16h a  = load_frag(lP,  wr * 16 + l15, 64, kc, lane);
      v16h b0 = load_frag(lVt, wc0 * 16 + l15, 64, kc, lane);
      v16h b1 = load_frag(lVt, (wc0 + 2) * 16 + l15, 64, kc, lane);
      o0 = wmma32(a, b0, o0);
      o1 = wmma32(a, b1, o1);
    }
    __syncthreads();
  }
  { int rb = wr * 16 + ((lane & 16) ? 8 : 0);
    #pragma unroll
    for (int i = 0; i < 8; ++i) {
      float inv = 1.f / lrow[rb + i];
      int row = qt * 64 + rb + i;
      long long base = ((long long)b * S + row) * D + hI * 64;
      Og[base + wc0 * 16 + l15]      = o0[i] * inv;
      Og[base + wc0 * 16 + 32 + l15] = o1[i] * inv;
    }
  }
}

// ---------------------------------------------------------------------------
// SSM scan: h_t = h_{t-1} @ A + xB_t. One block (64 threads) per batch.
// ---------------------------------------------------------------------------
__global__ __launch_bounds__(64)
void ssm_scan(const float* __restrict__ xB, const float* __restrict__ Amat,
              float* __restrict__ hfin)
{
  const int b = blockIdx.x, tid = threadIdx.x;
  __shared__ float sA[64 * 64];
  __shared__ float hbuf[2][64];
  for (int r = 0; r < 64; ++r) sA[r * 64 + tid] = Amat[r * 64 + tid];
  hbuf[0][tid] = 0.f;
  __syncthreads();
  int cur = 0;
  const float* xb = xB + (long long)b * 1024 * 64;
  for (int t = 0; t < 1024; ++t) {
    float acc = xb[t * 64 + tid];
    const float* h = hbuf[cur];
    #pragma unroll 8
    for (int i = 0; i < 64; ++i) acc += h[i] * sA[i * 64 + tid];
    hbuf[cur ^ 1][tid] = acc;
    cur ^= 1;
    __syncthreads();
  }
  hfin[b * 64 + tid] = hbuf[cur][tid];
}

__global__ void hproj_k(const float* __restrict__ hf, const float* __restrict__ Wimp,
                        float* __restrict__ hp)
{
  int idx = blockIdx.x * blockDim.x + threadIdx.x;   // 4096
  int b = idx >> 10, d = idx & 1023;
  float acc = 0.f;
  #pragma unroll 8
  for (int e = 0; e < 64; ++e) acc += hf[b * 64 + e] * Wimp[d * 64 + e];
  hp[idx] = acc;
}

__global__ void ilog_k(const float* __restrict__ x, const float* __restrict__ hp,
                       float* __restrict__ il)
{
  int idx = blockIdx.x * blockDim.x + threadIdx.x;   // 4096 = b*1024+s
  int b = idx >> 10, s = idx & 1023;
  const float* xr = x + ((long long)b * 1024 + s) * 1024;
  const float* hr = hp + b * 1024;
  float acc = 0.f;
  for (int d = 0; d < 1024; ++d) acc += xr[d] * hr[d];
  il[idx] = acc;
}

__global__ __launch_bounds__(256)
void softmax_s(const float* __restrict__ il, float* __restrict__ imp)
{
  const int b = blockIdx.x, tid = threadIdx.x;
  __shared__ float red[256];
  float m = -3.0e38f;
  for (int s = tid; s < 1024; s += 256) m = fmaxf(m, il[b * 1024 + s]);
  red[tid] = m; __syncthreads();
  for (int o = 128; o > 0; o >>= 1) {
    if (tid < o) red[tid] = fmaxf(red[tid], red[tid + o]);
    __syncthreads();
  }
  float M = red[0]; __syncthreads();
  float sum = 0.f;
  for (int s = tid; s < 1024; s += 256) sum += __expf(il[b * 1024 + s] - M);
  red[tid] = sum; __syncthreads();
  for (int o = 128; o > 0; o >>= 1) {
    if (tid < o) red[tid] += red[tid + o];
    __syncthreads();
  }
  float Z = red[0];
  for (int s = tid; s < 1024; s += 256)
    imp[b * 1024 + s] = __expf(il[b * 1024 + s] - M) / Z;
}

__global__ void zero_k(float* p, int n) {
  int i = blockIdx.x * blockDim.x + threadIdx.x;
  if (i < n) p[i] = 0.f;
}

__global__ void nw_acc_k(const float* __restrict__ prefL, const float* __restrict__ imp,
                         float* __restrict__ nw)
{
  int idx = blockIdx.x * blockDim.x + threadIdx.x;   // 4096
  int b = idx >> 10;
  const float* pl = prefL + (long long)idx * 32;
  float mx = -3.0e38f;
  #pragma unroll
  for (int n = 0; n < 32; ++n) mx = fmaxf(mx, pl[n]);
  float e[32]; float sum = 0.f;
  #pragma unroll
  for (int n = 0; n < 32; ++n) { e[n] = __expf(pl[n] - mx); sum += e[n]; }
  float w = imp[idx] / sum;
  #pragma unroll
  for (int n = 0; n < 32; ++n) atomicAdd(&nw[b * 32 + n], w * e[n]);
}

__global__ void nw_norm_k(float* __restrict__ nw) {
  int t = threadIdx.x;                                // 128
  __shared__ float s[128];
  s[t] = nw[t]; __syncthreads();
  int b = t >> 5;
  float sum = 0.f;
  #pragma unroll
  for (int i = 0; i < 32; ++i) sum += s[b * 32 + i];
  nw[t] = s[t] / (sum + 1e-8f);
}

// out[b][idx] = sum_n nw[b][n] * W[n][idx], all 4 batches per thread
__global__ __launch_bounds__(256)
void wsum_k(const float* __restrict__ W, const float* __restrict__ nw,
            float* __restrict__ out, int E)
{
  __shared__ float snw[128];
  if (threadIdx.x < 128) snw[threadIdx.x] = nw[threadIdx.x];
  __syncthreads();
  int idx = blockIdx.x * 256 + threadIdx.x;
  if (idx >= E) return;
  float a0 = 0.f, a1 = 0.f, a2 = 0.f, a3 = 0.f;
  #pragma unroll
  for (int n = 0; n < 32; ++n) {
    float w = W[(long long)n * E + idx];
    a0 += snw[n]      * w;
    a1 += snw[32 + n] * w;
    a2 += snw[64 + n] * w;
    a3 += snw[96 + n] * w;
  }
  out[idx]                = a0;
  out[(long long)E + idx] = a1;
  out[2LL * E + idx]      = a2;
  out[3LL * E + idx]      = a3;
}

// ---------------------------------------------------------------------------
extern "C" void kernel_launch(void* const* d_in, const int* in_sizes, int n_in,
                              void* d_out, int out_size, void* d_ws, size_t ws_size,
                              hipStream_t stream) {
  (void)in_sizes; (void)n_in; (void)out_size; (void)ws_size;
  const int B = 4, S = 1024, D = 1024, R = 256, SD = 64;
  const float* x      = (const float*)d_in[0];
  const float* Amat   = (const float*)d_in[2];
  const float* B_ssm  = (const float*)d_in[3];
  const float* W_imp  = (const float*)d_in[4];
  const float* W_rout = (const float*)d_in[5];
  const float* cn     = (const float*)d_in[6];
  const float* exQ    = (const float*)d_in[7];
  const float* exK    = (const float*)d_in[8];
  const float* exV    = (const float*)d_in[9];
  const float* W_O    = (const float*)d_in[10];
  float* out = (float*)d_out;

  char* w = (char*)d_ws;
  size_t off = 0;
  auto alloc = [&](size_t bytes) -> void* {
    void* p = w + off;
    off = (off + bytes + 255) & ~(size_t)255;
    return p;
  };
  float*    xBbuf = (float*)alloc((size_t)B * S * SD * 4);
  float*    hfin  = (float*)alloc((size_t)B * SD * 4);
  float*    hp    = (float*)alloc((size_t)B * D * 4);
  float*    il    = (float*)alloc((size_t)B * S * 4);
  float*    imp   = (float*)alloc((size_t)B * S * 4);
  float*    prefL = (float*)alloc((size_t)B * S * 32 * 4);
  float*    nw    = (float*)alloc((size_t)B * 32 * 4);
  float*    sc    = (float*)alloc((size_t)B * D * R * 4);
  float*    eQ    = (float*)alloc((size_t)B * R * D * 4);
  float*    eK    = (float*)alloc((size_t)B * R * D * 4);
  float*    eV    = (float*)alloc((size_t)B * R * D * 4);
  float*    hlr   = (float*)alloc((size_t)B * S * R * 4);
  _Float16* Qh    = (_Float16*)alloc((size_t)B * S * D * 2);
  _Float16* Kh    = (_Float16*)alloc((size_t)B * S * D * 2);
  _Float16* Vh    = (_Float16*)alloc((size_t)B * S * D * 2);
  float*    attn  = (float*)alloc((size_t)B * S * D * 4);

  dim3 blk(256);
  // 1) xB = x @ B_ssm       [B][S,64]
  gemm_f16w<false, float><<<dim3(1, 16, B), blk, 0, stream>>>(
      x, B_ssm, xBbuf, S, SD, D, D, SD, SD, (long long)S * D, 0, (long long)S * SD);
  // 2) sequential SSM scan -> h_final
  ssm_scan<<<dim3(B), dim3(64), 0, stream>>>(xBbuf, Amat, hfin);
  // 3) h_proj = h_final @ W_imp^T
  hproj_k<<<dim3(16), blk, 0, stream>>>(hfin, W_imp, hp);
  // 4) importance logits + softmax over S
  ilog_k<<<dim3(16), blk, 0, stream>>>(x, hp, il);
  softmax_s<<<dim3(B), blk, 0, stream>>>(il, imp);
  // 5) router logits = x @ W_router^T  (BT: W_router is [32][1024])
  gemm_f16w<true, float><<<dim3(1, 16, B), blk, 0, stream>>>(
      x, W_rout, prefL, S, 32, D, D, D, 32, (long long)S * D, 0, (long long)S * 32);
  // 6) nw = normalize( sum_s imp * softmax_n(prefL) )
  zero_k<<<dim3(1), dim3(128), 0, stream>>>(nw, B * 32);
  nw_acc_k<<<dim3(16), blk, 0, stream>>>(prefL, imp, nw);
  nw_norm_k<<<dim3(1), dim3(128), 0, stream>>>(nw);
  // 7) weighted neuron banks
  wsum_k<<<dim3((D * R) / 256), blk, 0, stream>>>(cn,  nw, sc, D * R);
  wsum_k<<<dim3((R * D) / 256), blk, 0, stream>>>(exQ, nw, eQ, R * D);
  wsum_k<<<dim3((R * D) / 256), blk, 0, stream>>>(exK, nw, eK, R * D);
  wsum_k<<<dim3((R * D) / 256), blk, 0, stream>>>(exV, nw, eV, R * D);
  // 8) h = x @ sc[b]        [B][S,R]
  gemm_f16w<false, float><<<dim3(R / 64, 16, B), blk, 0, stream>>>(
      x, sc, hlr, S, R, D, D, R, R, (long long)S * D, (long long)D * R, (long long)S * R);
  // 9) Q/K/V = h @ e{Q,K,V}[b]   (f16 output)
  gemm_f16w<false, _Float16><<<dim3(16, 16, B), blk, 0, stream>>>(
      hlr, eQ, Qh, S, D, R, R, D, D, (long long)S * R, (long long)R * D, (long long)S * D);
  gemm_f16w<false, _Float16><<<dim3(16, 16, B), blk, 0, stream>>>(
      hlr, eK, Kh, S, D, R, R, D, D, (long long)S * R, (long long)R * D, (long long)S * D);
  gemm_f16w<false, _Float16><<<dim3(16, 16, B), blk, 0, stream>>>(
      hlr, eV, Vh, S, D, R, R, D, D, (long long)S * R, (long long)R * D, (long long)S * D);
  // 10) causal flash attention
  flash_attn<<<dim3(16, 16, B), blk, 0, stream>>>(Qh, Kh, Vh, attn);
  // 11) output = attn_out @ W_O^T  (BT: W_O is [1024][1024] row-major)
  gemm_f16w<true, float><<<dim3(16, 16, B), blk, 0, stream>>>(
      attn, W_O, out, S, D, D, D, D, D, (long long)S * D, 0, (long long)S * D);
}
